// NATLayer_2602750182045
// MI455X (gfx1250) — compile-verified, matrix operand order checked by reading, not run on hardware
//
#include <hip/hip_runtime.h>
#include <hip/hip_bf16.h>
#include <math.h>

// ---------------------------------------------------------------------------
// NAT transformer layer, MI455X (gfx1250, wave32).
//   LN1 -> QKV GEMM(WMMA) -> neighborhood attention -> proj GEMM(WMMA)+resid
//       -> LN2 -> FC1 GEMM(WMMA)+GELU -> FC2 GEMM(WMMA)+resid
// GEMMs run on v_wmma_f32_16x16x4_f32 with 2x4 (MxN) register blocking per
// wave: 8 WMMAs per 6 loads in the K-loop. Epilogues (bias/GELU/residual)
// fused into the GEMM stores so intermediates make one L2 round trip.
// ---------------------------------------------------------------------------

#define NTOK   12544          // B*Hs*Ws = 4*56*56
#define HS     56
#define WS     56
#define CDIM   128
#define NHEADS 4
#define HD     32
#define KWIN   7
#define C3     (3*CDIM)

typedef __attribute__((ext_vector_type(2)))  float  v2f;
typedef __attribute__((ext_vector_type(8)))  float  v8f;
typedef __attribute__((ext_vector_type(16))) __bf16 v16bf;

#if __has_builtin(__builtin_amdgcn_wmma_f32_16x16x4_f32)
#define HAVE_WMMA_F32 1
#else
#define HAVE_WMMA_F32 0
#endif

// ---------------------------------------------------------------------------
// GEMM: OUT[M,N] = A[M,K] @ W[N,K]^T + bias  (+GELU | +residual)
// Each wave owns a 32x64 output macro-tile = 2 M-tiles x 4 N-tiles (16x16).
// Grid: x = groups of 4 N-tiles (4 waves/block), y = 32-row M super-tile.
// All exits are wave-uniform so EXEC is all-1s at every WMMA (ISA req).
//   EPI = 0: bias    EPI = 1: bias + residual    EPI = 2: bias + exact GELU
// Requires: M % 32 == 0, N % 64 == 0, K % 4 == 0 (true for all calls here).
// ---------------------------------------------------------------------------
template<int EPI>
__global__ __launch_bounds__(128) void gemm_bias_epi(
    const float* __restrict__ A, const float* __restrict__ W,
    const float* __restrict__ bias, const float* __restrict__ resid,
    float* __restrict__ out, int M, int K, int N)
{
  const int ngroups = N >> 6;                              // N / 64
  const int ng = blockIdx.x * (blockDim.x >> 5) + (threadIdx.x >> 5);
  if (ng >= ngroups) return;                               // wave-uniform
  const int lane = threadIdx.x & 31;
  const int half = lane >> 4;                              // K sub-pair selector
  const int sub  = lane & 15;                              // A row / B col in tile
  const int m0   = blockIdx.y << 5;                        // 32-row super-tile
  const int n0   = ng << 6;                                // 64-col super-tile

  const float* __restrict__ a0 = A + (size_t)(m0 + sub) * K + half * 2;
  const float* __restrict__ a1 = a0 + (size_t)16 * K;
  const float* __restrict__ b0 = W + (size_t)(n0 + sub) * K + half * 2;
  const float* __restrict__ b1 = b0 + (size_t)16 * K;
  const float* __restrict__ b2 = b0 + (size_t)32 * K;
  const float* __restrict__ b3 = b0 + (size_t)48 * K;
  __builtin_prefetch(b0, 0, 1);                            // global_prefetch_b8
  __builtin_prefetch(b2, 0, 1);

  v8f acc[8] = {};                                         // [mi*4 + t]
#if HAVE_WMMA_F32
  // ISA f32 16x4 A/B layout: lanes 0-15 hold {K=k,k+1}, lanes 16-31 {K=k+2,k+3}
#pragma unroll 4
  for (int k = 0; k < K; k += 4) {
    const v2f A0 = *(const v2f*)(a0 + k);
    const v2f A1 = *(const v2f*)(a1 + k);
    const v2f B0 = *(const v2f*)(b0 + k);
    const v2f B1 = *(const v2f*)(b1 + k);
    const v2f B2 = *(const v2f*)(b2 + k);
    const v2f B3 = *(const v2f*)(b3 + k);
    acc[0] = __builtin_amdgcn_wmma_f32_16x16x4_f32(false, A0, false, B0, (short)0, acc[0], false, false);
    acc[1] = __builtin_amdgcn_wmma_f32_16x16x4_f32(false, A0, false, B1, (short)0, acc[1], false, false);
    acc[2] = __builtin_amdgcn_wmma_f32_16x16x4_f32(false, A0, false, B2, (short)0, acc[2], false, false);
    acc[3] = __builtin_amdgcn_wmma_f32_16x16x4_f32(false, A0, false, B3, (short)0, acc[3], false, false);
    acc[4] = __builtin_amdgcn_wmma_f32_16x16x4_f32(false, A1, false, B0, (short)0, acc[4], false, false);
    acc[5] = __builtin_amdgcn_wmma_f32_16x16x4_f32(false, A1, false, B1, (short)0, acc[5], false, false);
    acc[6] = __builtin_amdgcn_wmma_f32_16x16x4_f32(false, A1, false, B2, (short)0, acc[6], false, false);
    acc[7] = __builtin_amdgcn_wmma_f32_16x16x4_f32(false, A1, false, B3, (short)0, acc[7], false, false);
  }
#else
  // Fallback: bf16 K=32 WMMA (codegen-confirmed), fp32 accumulate.
  const float* ab[2] = { a0 - half * 2, a1 - half * 2 };
  const float* bb[4] = { b0 - half * 2, b1 - half * 2, b2 - half * 2, b3 - half * 2 };
  for (int k0 = 0; k0 < K; k0 += 32) {
    v16bf av[2], bv[4];
#pragma unroll
    for (int e = 0; e < 16; ++e) {
      const int kk = k0 + ((e >> 3) << 4) + (half << 3) + (e & 7);
      av[0][e] = (__bf16)ab[0][kk];  av[1][e] = (__bf16)ab[1][kk];
      bv[0][e] = (__bf16)bb[0][kk];  bv[1][e] = (__bf16)bb[1][kk];
      bv[2][e] = (__bf16)bb[2][kk];  bv[3][e] = (__bf16)bb[3][kk];
    }
#pragma unroll
    for (int mi = 0; mi < 2; ++mi)
#pragma unroll
      for (int t = 0; t < 4; ++t)
        acc[mi * 4 + t] = __builtin_amdgcn_wmma_f32_16x16x32_bf16(
            false, av[mi], false, bv[t], (short)0, acc[mi * 4 + t], false, false);
  }
#endif

  // C/D layout: VGPR i -> row = tile_m + half*8 + i, col = tile_n + sub
#pragma unroll
  for (int mi = 0; mi < 2; ++mi) {
#pragma unroll
    for (int t = 0; t < 4; ++t) {
      const int col = n0 + (t << 4) + sub;
      const float bc = bias[col];
      const v8f a = acc[mi * 4 + t];
#pragma unroll
      for (int i = 0; i < 8; ++i) {
        const int row = m0 + (mi << 4) + (half << 3) + i;
        float v = a[i] + bc;
        if (EPI == 2) v = 0.5f * v * (1.0f + erff(v * 0.70710678118654752f));
        if (EPI == 1) v += resid[(size_t)row * N + col];
        out[(size_t)row * N + col] = v;
      }
    }
  }
}

// ---------------------------------------------------------------------------
// LayerNorm over C=128, one wave per token row (float4 per lane + shfl tree).
// ---------------------------------------------------------------------------
__global__ __launch_bounds__(256) void layernorm_k(
    const float* __restrict__ x, const float* __restrict__ w,
    const float* __restrict__ b, float* __restrict__ y, int rows)
{
  const int row = blockIdx.x * (blockDim.x >> 5) + (threadIdx.x >> 5);
  if (row >= rows) return;
  const int lane = threadIdx.x & 31;
  const float4 v = ((const float4*)(x + (size_t)row * CDIM))[lane];
  float s = v.x + v.y + v.z + v.w;
  for (int off = 16; off > 0; off >>= 1) s += __shfl_xor(s, off, 32);
  const float mu = s * (1.0f / CDIM);
  const float dx = v.x - mu, dy = v.y - mu, dz = v.z - mu, dw = v.w - mu;
  float s2 = dx * dx + dy * dy + dz * dz + dw * dw;
  for (int off = 16; off > 0; off >>= 1) s2 += __shfl_xor(s2, off, 32);
  const float rstd = rsqrtf(s2 * (1.0f / CDIM) + 1e-5f);
  const int c = lane * 4;
  float4 o;
  o.x = dx * rstd * w[c + 0] + b[c + 0];
  o.y = dy * rstd * w[c + 1] + b[c + 1];
  o.z = dz * rstd * w[c + 2] + b[c + 2];
  o.w = dw * rstd * w[c + 3] + b[c + 3];
  ((float4*)(y + (size_t)row * CDIM))[lane] = o;
}

// ---------------------------------------------------------------------------
// Neighborhood attention (7x7 window, clipped) + relative position bias.
// One wave per (token, head); lane = head-dim d (HD == 32 == wave32).
// Scores via shfl reductions; 49 attention weights staged in a per-wave LDS
// slab (intra-wave DS ops are in-order on CDNA5 -> no barrier needed).
// ---------------------------------------------------------------------------
__global__ __launch_bounds__(256) void nat_attn_k(
    const float* __restrict__ qkv, const float* __restrict__ rpb,
    float* __restrict__ out)
{
  __shared__ float smem[8][64];
  const int gw   = blockIdx.x * (blockDim.x >> 5) + (threadIdx.x >> 5);
  const int head = gw & (NHEADS - 1);
  const int tok  = gw >> 2;
  if (tok >= NTOK) return;
  const int lane = threadIdx.x & 31;
  float* sc = smem[threadIdx.x >> 5];

  const int w = tok % WS;
  const int h = (tok / WS) % HS;
  const int b = tok / (WS * HS);
  const int sh = min(max(h - KWIN / 2, 0), HS - KWIN);
  const int sw = min(max(w - KWIN / 2, 0), WS - KWIN);

  const float q = qkv[(size_t)tok * C3 + head * HD + lane] * 0.17677669529663687f; // hd^-0.5

  // --- scores[49] = q . k_win + rpb ---
  for (int ij = 0; ij < KWIN * KWIN; ++ij) {
    const int i = ij / KWIN, j = ij - i * KWIN;
    const int kh = sh + i, kw = sw + j;
    const size_t kidx = ((size_t)((b * HS + kh) * WS + kw)) * C3 + CDIM + head * HD + lane;
    float p = q * qkv[kidx];
    for (int off = 16; off > 0; off >>= 1) p += __shfl_xor(p, off, 32);
    if (lane == 0) {
      const int bh = kh - h + (KWIN - 1);
      const int bw = kw - w + (KWIN - 1);
      sc[ij] = p + rpb[(head * (2 * KWIN - 1) + bh) * (2 * KWIN - 1) + bw];
    }
  }

  // --- softmax over 49, 2 scores per lane ---
  const float s0 = sc[lane];
  const float s1 = (lane < 17) ? sc[32 + lane] : -3.4e38f;
  float m = fmaxf(s0, s1);
  for (int off = 16; off > 0; off >>= 1) m = fmaxf(m, __shfl_xor(m, off, 32));
  const float e0 = expf(s0 - m);
  const float e1 = (lane < 17) ? expf(s1 - m) : 0.0f;
  float ssum = e0 + e1;
  for (int off = 16; off > 0; off >>= 1) ssum += __shfl_xor(ssum, off, 32);
  const float inv = 1.0f / ssum;
  sc[lane] = e0 * inv;
  if (lane < 17) sc[32 + lane] = e1 * inv;

  // --- out_d = sum_ij attn[ij] * v[ij][d] ---
  float acc = 0.0f;
  for (int ij = 0; ij < KWIN * KWIN; ++ij) {
    const int i = ij / KWIN, j = ij - i * KWIN;
    const size_t vidx =
        ((size_t)((b * HS + sh + i) * WS + sw + j)) * C3 + 2 * CDIM + head * HD + lane;
    acc += sc[ij] * qkv[vidx];
  }
  out[(size_t)tok * CDIM + head * HD + lane] = acc;
}

// ---------------------------------------------------------------------------
extern "C" void kernel_launch(void* const* d_in, const int* in_sizes, int n_in,
                              void* d_out, int out_size, void* d_ws, size_t ws_size,
                              hipStream_t stream)
{
  const float* x      = (const float*)d_in[0];
  const float* n1w    = (const float*)d_in[1];
  const float* n1b    = (const float*)d_in[2];
  const float* qkv_w  = (const float*)d_in[3];
  const float* qkv_b  = (const float*)d_in[4];
  const float* rpb    = (const float*)d_in[5];
  const float* proj_w = (const float*)d_in[6];
  const float* proj_b = (const float*)d_in[7];
  const float* n2w    = (const float*)d_in[8];
  const float* n2b    = (const float*)d_in[9];
  const float* fc1_w  = (const float*)d_in[10];
  const float* fc1_b  = (const float*)d_in[11];
  const float* fc2_w  = (const float*)d_in[12];
  const float* fc2_b  = (const float*)d_in[13];
  float* out = (float*)d_out;

  // Workspace (floats): y/z (N*128) | qkv->h1 (N*512) | attn (N*128) | x2 (N*128)
  // Total ~45 MB; everything we read is written first each call (deterministic).
  float* ws   = (float*)d_ws;
  float* y    = ws;
  float* big  = y    + (size_t)NTOK * CDIM;
  float* attn = big  + (size_t)NTOK * 512;
  float* x2   = attn + (size_t)NTOK * CDIM;

  (void)in_sizes; (void)n_in; (void)out_size; (void)ws_size;

  const int mblk = NTOK / 32;                       // 392 super-tiles of 32 rows

  // 1. y = LN1(x)
  layernorm_k<<<NTOK / 8, 256, 0, stream>>>(x, n1w, n1b, y, NTOK);

  // 2. qkv = y @ qkv_w^T + qkv_b          (M=12544, K=128, N=384; 6 N-groups)
  gemm_bias_epi<0><<<dim3((6 + 3) / 4, mblk), 128, 0, stream>>>(
      y, qkv_w, qkv_b, nullptr, big, NTOK, CDIM, 384);

  // 3. attn = NAT(qkv, rpb)
  nat_attn_k<<<(NTOK * NHEADS) / 8, 256, 0, stream>>>(big, rpb, attn);

  // 4. x2 = x + attn @ proj_w^T + proj_b  (M=12544, K=128, N=128; 2 N-groups)
  gemm_bias_epi<1><<<dim3(1, mblk), 128, 0, stream>>>(
      attn, proj_w, proj_b, x, x2, NTOK, CDIM, CDIM);

  // 5. y = LN2(x2)
  layernorm_k<<<NTOK / 8, 256, 0, stream>>>(x2, n2w, n2b, y, NTOK);

  // 6. h1 = GELU(y @ fc1_w^T + fc1_b)     (M=12544, K=128, N=512; 8 N-groups)
  gemm_bias_epi<2><<<dim3(2, mblk), 128, 0, stream>>>(
      y, fc1_w, fc1_b, nullptr, big, NTOK, CDIM, 512);

  // 7. out = x2 + h1 @ fc2_w^T + fc2_b    (M=12544, K=512, N=128; 2 N-groups)
  gemm_bias_epi<1><<<dim3(1, mblk), 128, 0, stream>>>(
      big, fc2_w, fc2_b, x2, out, NTOK, 512, CDIM);
}
